// EdgeAttrGAT_16106127360273
// MI455X (gfx1250) — compile-verified, matrix operand order checked by reading, not run on hardware
//
#include <hip/hip_runtime.h>
#include <hip/hip_bf16.h>
#include <math.h>

// ---------------- problem constants ----------------
constexpr int NN = 10000;       // nodes
constexpr int EE = 160000;      // edges
constexpr int NF = 5;           // node feats
constexpr int EF = 3;           // edge feats
constexpr int HH = 8;           // heads
constexpr int CC = 128;         // per-head channels
constexpr int HC = HH * CC;     // 1024
constexpr int GG = 16;          // graphs
constexpr int E2 = EE + NN;     // edges + self loops

// ---------------- WMMA types ----------------
typedef __bf16 bf16x16 __attribute__((ext_vector_type(16)));
typedef float  f32x8   __attribute__((ext_vector_type(8)));

union FragAB { bf16x16 v; uint4 q[2]; };

__device__ __forceinline__ unsigned short f2bf(float f) {
    unsigned int u = __float_as_uint(f);
    unsigned int r = u + 0x7fffu + ((u >> 16) & 1u);   // round-to-nearest-even
    return (unsigned short)(r >> 16);
}

__device__ __forceinline__ unsigned int f2bf_pk(float lo, float hi) {
    return (unsigned int)f2bf(lo) | ((unsigned int)f2bf(hi) << 16);
}

__device__ __forceinline__ float atomicMaxF(float* addr, float value) {
    // sign-split trick: positive -> signed int max, negative -> unsigned min
    if (value >= 0.0f)
        return __int_as_float(atomicMax((int*)addr, __float_as_int(value)));
    return __uint_as_float(atomicMin((unsigned int*)addr, __float_as_uint(value)));
}

__device__ __forceinline__ void edge_info(int e, const int* __restrict__ ei,
                                          const float* __restrict__ edge_attr,
                                          const float* __restrict__ loop_attr,
                                          int& s, int& d, const float*& eap) {
    if (e < EE) { s = ei[e]; d = ei[EE + e]; eap = edge_attr + (size_t)e * EF; }
    else        { s = d = e - EE;            eap = loop_attr + (size_t)(e - EE) * EF; }
}

// ---------------- generic fill ----------------
__global__ void k_fill(float* __restrict__ p, float v, int n) {
    int i = blockIdx.x * blockDim.x + threadIdx.x;
    if (i < n) p[i] = v;
}

// ---------------- self-loop attr: mean of incoming edge_attr ----------------
__global__ void k_loop_accum(const int* __restrict__ ei, const float* __restrict__ ea,
                             float* __restrict__ loop_sum, float* __restrict__ cnt) {
    int e = blockIdx.x * blockDim.x + threadIdx.x;
    if (e >= EE) return;
    int d = ei[EE + e];
    atomicAdd(&cnt[d], 1.0f);
    #pragma unroll
    for (int f = 0; f < EF; ++f)
        atomicAdd(&loop_sum[d * EF + f], ea[(size_t)e * EF + f]);
}

__global__ void k_loop_final(float* __restrict__ loop_sum, const float* __restrict__ cnt) {
    int n = blockIdx.x * blockDim.x + threadIdx.x;
    if (n >= NN) return;
    float c = fmaxf(cnt[n], 1.0f);
    #pragma unroll
    for (int f = 0; f < EF; ++f) loop_sum[n * EF + f] /= c;
}

// ---------------- we_a[f][h] = sum_c We[f, h*C+c] * ae[h,c] ----------------
__global__ void k_wea(const float* __restrict__ We, const float* __restrict__ ae,
                      float* __restrict__ wea) {
    int t = threadIdx.x;
    if (t >= EF * HH) return;
    int f = t / HH, h = t % HH;
    float s = 0.0f;
    for (int c = 0; c < CC; ++c)
        s += We[(size_t)f * HC + h * CC + c] * ae[h * CC + c];
    wea[f * HH + h] = s;
}

// ---------------- layer-1 GEMM (K=5, VALU) ----------------
__global__ void k_gemm_smallk(const float* __restrict__ x, const float* __restrict__ W,
                              float* __restrict__ xl) {
    int idx = blockIdx.x * blockDim.x + threadIdx.x;
    if (idx >= NN * HC) return;
    int n = idx >> 10, j = idx & 1023;
    float s = 0.0f;
    #pragma unroll
    for (int f = 0; f < NF; ++f)
        s += x[n * NF + f] * W[(size_t)f * HC + j];
    xl[idx] = s;
}

// ---------------- bf16 WMMA GEMM: xl[N,1024] = act[N,1024] @ W[1024,1024] ----------------
// block = 256 threads (8 waves), block tile 64x128, K panel = 32
// wave -> m_sub (16 rows) x 4 n-subtiles (64 cols) => 4 WMMAs per K-step per wave
__global__ void __launch_bounds__(256)
k_gemm_wmma(const float* __restrict__ act, const float* __restrict__ Wm,
            float* __restrict__ xl) {
    constexpr int BM = 64, BN = 128, BK = 32;
    __shared__ __align__(16) unsigned short As[BM * BK];  // [row][k]
    __shared__ __align__(16) unsigned short Bs[BN * BK];  // [col][k]

    const int t       = threadIdx.x;
    const int colBase = blockIdx.x * BN;
    const int rowBase = blockIdx.y * BM;

    const int wave = t >> 5;
    const int lane = t & 31;
    const int lo   = lane & 15;
    const int hi   = lane >> 4;
    const int m_sub  = wave & 3;           // 0..3  (16-row subtile)
    const int n_base = (wave >> 2) * 4;    // 0 or 4 (first of 4 16-col subtiles)

    // A-stage coords: 4 threads per row, 8 consecutive k each
    const int ar = t >> 2;                 // 0..63
    const int ak = (t & 3) * 8;            // 0,8,16,24
    // B-stage coords: thread owns k-pair (kp, kp+1) x 8 consecutive cols
    const int bcg = t >> 4;                // 0..15 -> col group
    const int bc  = bcg * 8;
    const int kp  = (t & 15) * 2;          // 0,2,..,30

    f32x8 acc[4];
    #pragma unroll
    for (int j = 0; j < 4; ++j) acc[j] = (f32x8){0.f,0.f,0.f,0.f,0.f,0.f,0.f,0.f};

    for (int k0 = 0; k0 < HC; k0 += BK) {
        // ---- stage A (f32 -> bf16), rows of act ----
        const int grow = rowBase + ar;
        if (grow < NN) {
            const float* ap = act + (size_t)grow * HC + k0 + ak;
            #pragma unroll
            for (int i = 0; i < 8; ++i) As[ar * BK + ak + i] = f2bf(ap[i]);
        } else {
            #pragma unroll
            for (int i = 0; i < 8; ++i) As[ar * BK + ak + i] = 0;
        }
        // ---- stage B transposed+packed: Bs[col][k], k-pair packed as one b32 ----
        {
            const float* w0 = Wm + (size_t)(k0 + kp) * HC + colBase + bc;
            const float* w1 = w0 + HC;
            #pragma unroll
            for (int i = 0; i < 8; ++i) {
                unsigned int u = f2bf_pk(w0[i], w1[i]);
                *reinterpret_cast<unsigned int*>(&Bs[(bc + i) * BK + kp]) = u;
            }
        }
        __syncthreads();

        FragAB a;
        {   // A fragment: row = m_sub*16+lo ; K pattern per 16-bit WMMA layout
            int base = (m_sub * 16 + lo) * BK + hi * 8;
            a.q[0] = *reinterpret_cast<const uint4*>(&As[base]);
            a.q[1] = *reinterpret_cast<const uint4*>(&As[base + 16]);
        }
        #pragma unroll
        for (int j = 0; j < 4; ++j) {
            FragAB b;
            int base = ((n_base + j) * 16 + lo) * BK + hi * 8;
            b.q[0] = *reinterpret_cast<const uint4*>(&Bs[base]);
            b.q[1] = *reinterpret_cast<const uint4*>(&Bs[base + 16]);
            acc[j] = __builtin_amdgcn_wmma_f32_16x16x32_bf16(false, a.v, false, b.v,
                                                             (short)0, acc[j], false, false);
        }
        __syncthreads();
    }

    // store: C VGPR r -> row = hi*8 + r, col = lo (per 32-bit C/D layout)
    #pragma unroll
    for (int j = 0; j < 4; ++j) {
        #pragma unroll
        for (int r = 0; r < 8; ++r) {
            int row = rowBase + m_sub * 16 + hi * 8 + r;
            if (row < NN)
                xl[(size_t)row * HC + colBase + (n_base + j) * 16 + lo] = acc[j][r];
        }
    }
}

// ---------------- a_src / a_dst: per (n,h) dot over C, one wave each ----------------
__global__ void k_attn(const float* __restrict__ xl, const float* __restrict__ a_s,
                       const float* __restrict__ a_d, float* __restrict__ asrc,
                       float* __restrict__ adst) {
    int wg = blockIdx.x * 8 + (threadIdx.x >> 5);
    if (wg >= NN * HH) return;
    int lane = threadIdx.x & 31;
    int n = wg / HH, h = wg % HH;
    const float4 xv = *reinterpret_cast<const float4*>(&xl[(size_t)n * HC + h * CC + lane * 4]);
    const float4 sv = *reinterpret_cast<const float4*>(&a_s[h * CC + lane * 4]);
    const float4 dv = *reinterpret_cast<const float4*>(&a_d[h * CC + lane * 4]);
    float ps = xv.x * sv.x + xv.y * sv.y + xv.z * sv.z + xv.w * sv.w;
    float pd = xv.x * dv.x + xv.y * dv.y + xv.z * dv.z + xv.w * dv.w;
    #pragma unroll
    for (int m = 16; m >= 1; m >>= 1) {
        ps += __shfl_xor(ps, m, 32);
        pd += __shfl_xor(pd, m, 32);
    }
    if (lane == 0) { asrc[wg] = ps; adst[wg] = pd; }
}

// ---------------- alpha = leaky_relu(asrc[s]+adst[d]+ea.wea) ; segment max ----------------
__global__ void k_edge_alpha(const int* __restrict__ ei, const float* __restrict__ ea,
                             const float* __restrict__ loop_attr,
                             const float* __restrict__ asrc, const float* __restrict__ adst,
                             const float* __restrict__ wea,
                             float* __restrict__ alpha, float* __restrict__ m) {
    int idx = blockIdx.x * blockDim.x + threadIdx.x;
    if (idx >= E2 * HH) return;
    int e = idx / HH, h = idx - e * HH;
    int s, d; const float* eap;
    edge_info(e, ei, ea, loop_attr, s, d, eap);
    float ed = eap[0] * wea[0 * HH + h] + eap[1] * wea[1 * HH + h] + eap[2] * wea[2 * HH + h];
    float al = asrc[s * HH + h] + adst[d * HH + h] + ed;
    al = (al > 0.0f) ? al : 0.2f * al;
    alpha[idx] = al;
    atomicMaxF(&m[d * HH + h], al);
}

// ---------------- ex = exp(alpha - m[dst]) ; segment sum ----------------
__global__ void k_edge_exp(const int* __restrict__ ei,
                           const float* __restrict__ m, float* __restrict__ alpha,
                           float* __restrict__ den) {
    int idx = blockIdx.x * blockDim.x + threadIdx.x;
    if (idx >= E2 * HH) return;
    int e = idx / HH, h = idx - e * HH;
    int d = (e < EE) ? ei[EE + e] : (e - EE);
    float ex = expf(alpha[idx] - m[d * HH + h]);
    alpha[idx] = ex;
    atomicAdd(&den[d * HH + h], ex);
}

// ---------------- out[dst] += (ex/den) * xl[src] : one block per edge ----------------
__global__ void __launch_bounds__(256)
k_aggregate(const int* __restrict__ ei, const float* __restrict__ xl,
            const float* __restrict__ alpha, const float* __restrict__ den,
            float* __restrict__ out) {
    int e = blockIdx.x;
    int t = threadIdx.x;
    int s, d;
    if (e < EE) { s = ei[e]; d = ei[EE + e]; }
    else        { s = d = e - EE; }
    int h = t >> 5;                 // 32 threads per head
    int j = t * 4;                  // 4 channels per thread
    float w = alpha[(size_t)e * HH + h] / (den[d * HH + h] + 1e-16f);
    const float4 xv = *reinterpret_cast<const float4*>(&xl[(size_t)s * HC + j]);
    float* op = &out[(size_t)d * HC + j];
    atomicAdd(op + 0, w * xv.x);
    atomicAdd(op + 1, w * xv.y);
    atomicAdd(op + 2, w * xv.z);
    atomicAdd(op + 3, w * xv.w);
}

// ---------------- bias + elu (concat layers), in place ----------------
__global__ void k_final_concat(float* __restrict__ act, const float* __restrict__ b) {
    int idx = blockIdx.x * blockDim.x + threadIdx.x;
    if (idx >= NN * HC) return;
    float v = act[idx] + b[idx & 1023];
    act[idx] = (v > 0.0f) ? v : (expf(v) - 1.0f);
}

// ---------------- head-mean + bias + elu (layer 4) ----------------
__global__ void k_final_mean(const float* __restrict__ out, const float* __restrict__ b,
                             float* __restrict__ h4) {
    int idx = blockIdx.x * blockDim.x + threadIdx.x;
    if (idx >= NN * CC) return;
    int n = idx / CC, c = idx - n * CC;
    float s = 0.0f;
    #pragma unroll
    for (int h = 0; h < HH; ++h) s += out[(size_t)n * HC + h * CC + c];
    float v = s * (1.0f / HH) + b[c];
    h4[idx] = (v > 0.0f) ? v : (expf(v) - 1.0f);
}

// ---------------- pooling ----------------
__global__ void k_pool_cnt(const int* __restrict__ batch, float* __restrict__ cntg) {
    int n = blockIdx.x * blockDim.x + threadIdx.x;
    if (n < NN) atomicAdd(&cntg[batch[n]], 1.0f);
}

__global__ void k_pool_sum(const int* __restrict__ batch, const float* __restrict__ h4,
                           float* __restrict__ pooled) {
    int idx = blockIdx.x * blockDim.x + threadIdx.x;
    if (idx >= NN * CC) return;
    int n = idx / CC, c = idx - n * CC;
    atomicAdd(&pooled[batch[n] * CC + c], h4[idx]);
}

__global__ void k_final_out(const float* __restrict__ pooled, const float* __restrict__ cntg,
                            const float* __restrict__ lin_w, const float* __restrict__ lin_b,
                            float* __restrict__ outv) {
    int g = threadIdx.x >> 5;
    int lane = threadIdx.x & 31;
    if (g >= GG) return;
    float s = 0.0f;
    #pragma unroll
    for (int i = 0; i < 4; ++i) {
        int c = lane * 4 + i;
        s += pooled[g * CC + c] * lin_w[c];
    }
    #pragma unroll
    for (int m = 16; m >= 1; m >>= 1) s += __shfl_xor(s, m, 32);
    if (lane == 0) outv[g] = s / fmaxf(cntg[g], 1.0f) + lin_b[0];
}

// ======================= host launcher =======================
extern "C" void kernel_launch(void* const* d_in, const int* in_sizes, int n_in,
                              void* d_out, int out_size, void* d_ws, size_t ws_size,
                              hipStream_t stream) {
    (void)in_sizes; (void)n_in; (void)out_size; (void)ws_size;

    const float* x   = (const float*)d_in[0];
    const int*   ei  = (const int*)d_in[1];
    const float* ea  = (const float*)d_in[2];
    const int*   bat = (const int*)d_in[3];
    const float *W[4], *We[4], *As[4], *Ad[4], *Ae[4], *B[4];
    for (int l = 0; l < 4; ++l) {
        int base = 4 + l * 6;
        W[l]  = (const float*)d_in[base + 0];
        We[l] = (const float*)d_in[base + 1];
        As[l] = (const float*)d_in[base + 2];
        Ad[l] = (const float*)d_in[base + 3];
        Ae[l] = (const float*)d_in[base + 4];
        B[l]  = (const float*)d_in[base + 5];
    }
    const float* lin_w = (const float*)d_in[28];
    const float* lin_b = (const float*)d_in[29];
    float* outv = (float*)d_out;

    // ---- workspace carve (floats) ----
    float*  ws  = (float*)d_ws;
    size_t  off = 0;
    auto alloc = [&](size_t n) { float* p = ws + off; off += n; return p; };
    float* xl      = alloc((size_t)NN * HC);
    float* actA    = alloc((size_t)NN * HC);
    float* actB    = alloc((size_t)NN * HC);
    float* loopat  = alloc((size_t)NN * EF);   // sum -> mean in place
    float* cnt     = alloc(NN);
    float* asrc    = alloc((size_t)NN * HH);
    float* adst    = alloc((size_t)NN * HH);
    float* mmax    = alloc((size_t)NN * HH);
    float* den     = alloc((size_t)NN * HH);
    float* alpha   = alloc((size_t)E2 * HH);
    float* wea     = alloc(EF * HH);
    float* pooled  = alloc(GG * CC);
    float* cntg    = alloc(GG);

    const int T = 256;
    auto blocks = [](long n, int t) { return (int)((n + t - 1) / t); };

    // ---- self-loop edge attr (mean of incoming) ----
    k_fill<<<blocks(NN, T), T, 0, stream>>>(cnt, 0.0f, NN);
    k_fill<<<blocks(NN * EF, T), T, 0, stream>>>(loopat, 0.0f, NN * EF);
    k_loop_accum<<<blocks(EE, T), T, 0, stream>>>(ei, ea, loopat, cnt);
    k_loop_final<<<blocks(NN, T), T, 0, stream>>>(loopat, cnt);

    const float NEG_INF = -__builtin_huge_valf();
    float* act_in = nullptr;     // layer input (fp32 [N,1024]); layer 1 uses x

    for (int l = 0; l < 4; ++l) {
        float* act_out = (l % 2 == 0) ? actA : actB;

        // tiny contraction: wea[f][h]
        k_wea<<<1, 32, 0, stream>>>(We[l], Ae[l], wea);

        // GEMM -> xl
        if (l == 0) {
            k_gemm_smallk<<<blocks((long)NN * HC, T), T, 0, stream>>>(x, W[0], xl);
        } else {
            dim3 g(HC / 128, (NN + 63) / 64);
            k_gemm_wmma<<<g, T, 0, stream>>>(act_in, W[l], xl);
        }

        // per-node attention terms
        k_attn<<<blocks((long)NN * HH, 8), T, 0, stream>>>(xl, As[l], Ad[l], asrc, adst);

        // segment softmax over dst
        k_fill<<<blocks(NN * HH, T), T, 0, stream>>>(mmax, NEG_INF, NN * HH);
        k_fill<<<blocks(NN * HH, T), T, 0, stream>>>(den, 0.0f, NN * HH);
        k_edge_alpha<<<blocks((long)E2 * HH, T), T, 0, stream>>>(ei, ea, loopat, asrc, adst,
                                                                 wea, alpha, mmax);
        k_edge_exp<<<blocks((long)E2 * HH, T), T, 0, stream>>>(ei, mmax, alpha, den);

        // weighted aggregation into act_out
        k_fill<<<blocks((long)NN * HC, T), T, 0, stream>>>(act_out, 0.0f, NN * HC);
        k_aggregate<<<E2, T, 0, stream>>>(ei, xl, alpha, den, act_out);

        if (l < 3) {
            k_final_concat<<<blocks((long)NN * HC, T), T, 0, stream>>>(act_out, B[l]);
            act_in = act_out;
        } else {
            // head mean -> h4 (reuse xl buffer, free after aggregation)
            k_final_mean<<<blocks((long)NN * CC, T), T, 0, stream>>>(act_out, B[3], xl);
        }
    }
    float* h4 = xl;

    // ---- graph mean pool + linear ----
    k_fill<<<1, T, 0, stream>>>(pooled, 0.0f, GG * CC);
    k_fill<<<1, 32, 0, stream>>>(cntg, 0.0f, GG);
    k_pool_cnt<<<blocks(NN, T), T, 0, stream>>>(bat, cntg);
    k_pool_sum<<<blocks((long)NN * CC, T), T, 0, stream>>>(bat, h4, pooled);
    k_final_out<<<1, GG * 32, 0, stream>>>(pooled, cntg, lin_w, lin_b, outv);
}
